// Raindrop_v2_56796647522441
// MI455X (gfx1250) — matrix operand order repaired and checked by reading, not compiled, not run on hardware
//
#include <hip/hip_runtime.h>
#include <hip/hip_bf16.h>
#include <math.h>

typedef __attribute__((ext_vector_type(16))) _Float16 v16h;
typedef __attribute__((ext_vector_type(8)))  float    v8f;
typedef __attribute__((ext_vector_type(4)))  unsigned v4u;

union H16 {
  v16h h;
  v4u  q[2];
  unsigned u[8];
  _Float16 e[16];
};

__device__ inline v8f wmma_f16(const H16& a, const H16& b, v8f c) {
  return __builtin_amdgcn_wmma_f32_16x16x32_f16(false, a.h, false, b.h, (short)0, c, false, false);
}

#define GM_OUT_F32 0
#define GM_OUT_F16 1
#define GM_OUT_QKV 2

template <int MODE, bool RELU>
__device__ inline void store_tile(v8f acc, int mrBase, int half, int n, float bval,
                                  float* __restrict__ outF, _Float16* __restrict__ outH, int ldc,
                                  _Float16* __restrict__ qh, _Float16* __restrict__ kh,
                                  _Float16* __restrict__ vT, int T, int Bb)
{
#pragma unroll
  for (int vv = 0; vv < 8; ++vv) {
    int mr = mrBase + vv + (half << 3);   // C/D row layout
    float r = acc[vv] + bval;
    if (RELU) r = fmaxf(r, 0.f);
    if (MODE == GM_OUT_F32) {
      outF[(size_t)mr * ldc + n] = r;
    } else if (MODE == GM_OUT_F16) {
      outH[(size_t)mr * ldc + n] = (_Float16)r;
    } else { // QKV scatter: row mr = t*Bb+b, col n in [0,480)
      int t = mr / Bb, bb = mr % Bb;
      int which = n / 160, cc = n % 160;
      int hh = cc / 40, dd = cc % 40;
      int bh = bb * 4 + hh;
      if (which == 0)      qh[((size_t)bh * 224 + t) * 64 + dd] = (_Float16)r;
      else if (which == 1) kh[((size_t)bh * 224 + t) * 64 + dd] = (_Float16)r;
      else                 vT[((size_t)bh * 48 + dd) * 224 + t] = (_Float16)r;
    }
  }
}

// C[m][n] = act( sum_k A[m][k] * W[n][k] + bias[n] )
// A: (M x K) f16 row-major (lda == K, K % 32 == 0), W: (N x K) f16 row-major.
// M % 32 == 0, N % 32 == 0 (pre-padded). 32x32 register tile per wave:
// branch-free inner loop = 8x global_load_b128 -> 4x wmma.
template <int MODE, bool RELU>
__global__ __launch_bounds__(256) void gemm_wmma(
    const _Float16* __restrict__ A, int lda,
    const _Float16* __restrict__ W, int ldw,
    const float* __restrict__ bias,
    int M, int N, int K,
    float* __restrict__ outF, _Float16* __restrict__ outH, int ldc,
    _Float16* __restrict__ qh, _Float16* __restrict__ kh, _Float16* __restrict__ vT,
    int T, int Bb)
{
  int lane = threadIdx.x & 31;
  int wave = (blockIdx.x * blockDim.x + threadIdx.x) >> 5;
  int wavesTotal = (gridDim.x * blockDim.x) >> 5;
  int tilesN = N >> 5;
  int totalTiles = (M >> 5) * tilesN;
  int l15 = lane & 15;
  int half = lane >> 4;
  for (int tile = wave; tile < totalTiles; tile += wavesTotal) {
    int tm = tile / tilesN, tn = tile % tilesN;
    int m0 = (tm << 5) + l15;
    int n0 = (tn << 5) + l15;
    const _Float16* Arow0 = A + (size_t)m0 * lda + (half << 3);
    const _Float16* Arow1 = Arow0 + (size_t)16 * lda;
    const _Float16* Wrow0 = W + (size_t)n0 * ldw + (half << 3);
    const _Float16* Wrow1 = Wrow0 + (size_t)16 * ldw;
    v8f acc00 = {0.f,0.f,0.f,0.f,0.f,0.f,0.f,0.f};
    v8f acc01 = acc00, acc10 = acc00, acc11 = acc00;
    for (int ks = 0; ks < K; ks += 32) {
      H16 a0, a1, b0, b1;
      a0.q[0] = *(const v4u*)(Arow0 + ks);
      a0.q[1] = *(const v4u*)(Arow0 + ks + 16);
      a1.q[0] = *(const v4u*)(Arow1 + ks);
      a1.q[1] = *(const v4u*)(Arow1 + ks + 16);
      b0.q[0] = *(const v4u*)(Wrow0 + ks);
      b0.q[1] = *(const v4u*)(Wrow0 + ks + 16);
      b1.q[0] = *(const v4u*)(Wrow1 + ks);
      b1.q[1] = *(const v4u*)(Wrow1 + ks + 16);
      acc00 = wmma_f16(a0, b0, acc00);
      acc01 = wmma_f16(a0, b1, acc01);
      acc10 = wmma_f16(a1, b0, acc10);
      acc11 = wmma_f16(a1, b1, acc11);
    }
    float bv0 = bias[n0];
    float bv1 = bias[n0 + 16];
    int mb0 = tm << 5, mb1 = (tm << 5) + 16;
    store_tile<MODE, RELU>(acc00, mb0, half, n0,      bv0, outF, outH, ldc, qh, kh, vT, T, Bb);
    store_tile<MODE, RELU>(acc01, mb0, half, n0 + 16, bv1, outF, outH, ldc, qh, kh, vT, T, Bb);
    store_tile<MODE, RELU>(acc10, mb1, half, n0,      bv0, outF, outH, ldc, qh, kh, vT, T, Bb);
    store_tile<MODE, RELU>(acc11, mb1, half, n0 + 16, bv1, outF, outH, ldc, qh, kh, vT, T, Bb);
  }
}

// One workgroup per (batch, head).
// LDS: 224x224 f32 scores + 224x224 f16 probabilities (294 KB of the 320 KB WGP LDS).
__global__ __launch_bounds__(256) void attention_kernel(
    const _Float16* __restrict__ qh, const _Float16* __restrict__ kh,
    const _Float16* __restrict__ vT, const int* __restrict__ lengths,
    _Float16* __restrict__ oh, int T, int Bb)
{
  extern __shared__ char smem[];
  const int Tp = 224;
  float*    S = (float*)smem;                      // Tp*Tp f32
  _Float16* P = (_Float16*)(smem + Tp * Tp * 4);   // Tp*Tp f16
  int bh = blockIdx.x;
  int b = bh >> 2, h = bh & 3;
  int len = lengths[b];
  const _Float16* q = qh + (size_t)bh * Tp * 64;
  const _Float16* k = kh + (size_t)bh * Tp * 64;
  const _Float16* v = vT + (size_t)bh * 48 * Tp;
  int lane = threadIdx.x & 31;
  int wave = threadIdx.x >> 5;
  int l15 = lane & 15, half = lane >> 4;
  const float scale = 0.15811388300841898f; // 1/sqrt(40)

  // ---- S = (Q K^T)/sqrt(hd), masked; 32x32 tile per wave (7x7 tiles) ----
  for (int tile = wave; tile < 7 * 7; tile += 8) {
    int tm = tile / 7, tn = tile % 7;
    int m0 = (tm << 5) + l15;
    int n0 = (tn << 5) + l15;
    const _Float16* qrow0 = q + (size_t)m0 * 64 + (half << 3);
    const _Float16* qrow1 = qrow0 + 16 * 64;
    const _Float16* krow0 = k + (size_t)n0 * 64 + (half << 3);
    const _Float16* krow1 = krow0 + 16 * 64;
    v8f acc00 = {0.f,0.f,0.f,0.f,0.f,0.f,0.f,0.f};
    v8f acc01 = acc00, acc10 = acc00, acc11 = acc00;
#pragma unroll
    for (int ks = 0; ks < 64; ks += 32) {
      H16 a0, a1, b0, b1;
      a0.q[0] = *(const v4u*)(qrow0 + ks);
      a0.q[1] = *(const v4u*)(qrow0 + ks + 16);
      a1.q[0] = *(const v4u*)(qrow1 + ks);
      a1.q[1] = *(const v4u*)(qrow1 + ks + 16);
      b0.q[0] = *(const v4u*)(krow0 + ks);
      b0.q[1] = *(const v4u*)(krow0 + ks + 16);
      b1.q[0] = *(const v4u*)(krow1 + ks);
      b1.q[1] = *(const v4u*)(krow1 + ks + 16);
      acc00 = wmma_f16(a0, b0, acc00);
      acc01 = wmma_f16(a0, b1, acc01);
      acc10 = wmma_f16(a1, b0, acc10);
      acc11 = wmma_f16(a1, b1, acc11);
    }
    bool mask0 = (n0 >= len) || (n0 >= T);
    bool mask1 = (n0 + 16 >= len) || (n0 + 16 >= T);
#pragma unroll
    for (int vv = 0; vv < 8; ++vv) {
      int mr0 = (tm << 5) + vv + (half << 3);
      int mr1 = mr0 + 16;
      S[mr0 * Tp + n0]      = mask0 ? -1e30f : acc00[vv] * scale;
      S[mr0 * Tp + n0 + 16] = mask1 ? -1e30f : acc01[vv] * scale;
      S[mr1 * Tp + n0]      = mask0 ? -1e30f : acc10[vv] * scale;
      S[mr1 * Tp + n0 + 16] = mask1 ? -1e30f : acc11[vv] * scale;
    }
  }
  __syncthreads();

  // ---- wave-per-row softmax: f32 scores -> normalized f16 probabilities ----
  for (int r = wave; r < Tp; r += 8) {
    const float* row = S + r * Tp;
    float e[7];
    float mx = -1e30f;
#pragma unroll
    for (int j = 0; j < 7; ++j) { e[j] = row[lane + j * 32]; mx = fmaxf(mx, e[j]); }
#pragma unroll
    for (int off = 16; off > 0; off >>= 1) mx = fmaxf(mx, __shfl_xor(mx, off, 32));
    float sum = 0.f;
#pragma unroll
    for (int j = 0; j < 7; ++j) { e[j] = __expf(e[j] - mx); sum += e[j]; }
#pragma unroll
    for (int off = 16; off > 0; off >>= 1) sum += __shfl_xor(sum, off, 32);
    float inv = 1.f / sum;
    _Float16* prow = P + r * Tp;
#pragma unroll
    for (int j = 0; j < 7; ++j) prow[lane + j * 32] = (_Float16)(e[j] * inv);
  }
  __syncthreads();

  // ---- O = P @ V: 32x16 tile per wave (7 x 3 tiles), all b128 ----
  for (int tile = wave; tile < 7 * 3; tile += 8) {
    int tm = tile / 3, tn = tile % 3;
    int m0 = (tm << 5) + l15;
    int n = (tn << 4) + l15;
    const _Float16* prow0 = P + m0 * Tp + (half << 3);
    const _Float16* prow1 = prow0 + 16 * Tp;
    const _Float16* vrow = v + (size_t)n * Tp + (half << 3);
    v8f acc0 = {0.f,0.f,0.f,0.f,0.f,0.f,0.f,0.f};
    v8f acc1 = acc0;
    for (int ks = 0; ks < Tp; ks += 32) {
      H16 a0, a1, bm;
      a0.q[0] = *(const v4u*)(prow0 + ks);
      a0.q[1] = *(const v4u*)(prow0 + ks + 16);
      a1.q[0] = *(const v4u*)(prow1 + ks);
      a1.q[1] = *(const v4u*)(prow1 + ks + 16);
      bm.q[0] = *(const v4u*)(vrow + ks);
      bm.q[1] = *(const v4u*)(vrow + ks + 16);
      acc0 = wmma_f16(a0, bm, acc0);
      acc1 = wmma_f16(a1, bm, acc1);
    }
#pragma unroll
    for (int vv = 0; vv < 8; ++vv) {
      int mr0 = (tm << 5) + vv + (half << 3);
      int mr1 = mr0 + 16;
      if (n < 40) {
        if (mr0 < T) oh[((size_t)mr0 * Bb + b) * 160 + h * 40 + n] = (_Float16)acc0[vv];
        if (mr1 < T) oh[((size_t)mr1 * Bb + b) * 160 + h * 40 + n] = (_Float16)acc1[vv];
      }
    }
  }
}

// x = LN(x + delta) * s + b   (row width 160, one wave per row)
__global__ __launch_bounds__(256) void ln_kernel(
    float* __restrict__ x, const float* __restrict__ delta,
    const float* __restrict__ s, const float* __restrict__ bi,
    _Float16* __restrict__ xh, int rows)
{
  int wave = (blockIdx.x * blockDim.x + threadIdx.x) >> 5;
  int lane = threadIdx.x & 31;
  if (wave >= rows) return;
  size_t base = (size_t)wave * 160;
  float vals[5];
  float sum = 0.f, sq = 0.f;
#pragma unroll
  for (int j = 0; j < 5; ++j) {
    int c = lane + j * 32;
    float vv = x[base + c] + delta[base + c];
    vals[j] = vv; sum += vv; sq += vv * vv;
  }
#pragma unroll
  for (int off = 16; off > 0; off >>= 1) {
    sum += __shfl_xor(sum, off, 32);
    sq  += __shfl_xor(sq,  off, 32);
  }
  float mu = sum * (1.f / 160.f);
  float var = sq * (1.f / 160.f) - mu * mu;
  float rstd = rsqrtf(var + 1e-5f);
#pragma unroll
  for (int j = 0; j < 5; ++j) {
    int c = lane + j * 32;
    float vv = (vals[j] - mu) * rstd * s[c] + bi[c];
    x[base + c] = vv;
    xh[base + c] = (_Float16)vv;
  }
}

__global__ void f32_to_f16(const float* __restrict__ in, _Float16* __restrict__ out, int n) {
  int i = blockIdx.x * blockDim.x + threadIdx.x;
  if (i < n) out[i] = (_Float16)in[i];
}

// Pad-convert (N x K) f32 -> (Np x Kp) f16, zero-filled pads.
__global__ void f32_to_f16_pad(const float* __restrict__ in, _Float16* __restrict__ out,
                               int N, int K, int Np, int Kp) {
  int i = blockIdx.x * blockDim.x + threadIdx.x;
  if (i >= Np * Kp) return;
  int n = i / Kp, k = i % Kp;
  out[i] = (n < N && k < K) ? (_Float16)in[(size_t)n * K + k] : (_Float16)0.f;
}

__global__ void pad_bias(const float* __restrict__ in, float* __restrict__ out, int N, int Np) {
  int i = blockIdx.x * blockDim.x + threadIdx.x;
  if (i < Np) out[i] = (i < N) ? in[i] : 0.f;
}

// Xb[b][i][c] = relu(src[t][b][i] * R_u[i*4+k]) as f16, padded to 864 cols.
__global__ void prep_xb(const float* __restrict__ src, const float* __restrict__ Ru,
                        _Float16* __restrict__ xb, int T, int Bb) {
  int idx = blockIdx.x * blockDim.x + threadIdx.x;
  int total = Bb * 36 * 864;
  if (idx >= total) return;
  int b = idx / (36 * 864);
  int rem = idx % (36 * 864);
  int i = rem / 864;
  int c = rem % 864;
  float vv = 0.f;
  if (c < 860) {
    int t = c >> 2, k2 = c & 3;
    float obs = src[((size_t)t * Bb + b) * 72 + i];
    vv = fmaxf(obs * Ru[i * 4 + k2], 0.f);
  }
  xb[idx] = (_Float16)vv;
}

__global__ void emb_kernel(const float* __restrict__ stat, const float* __restrict__ ew,
                           const float* __restrict__ eb, float* __restrict__ emb) {
  int idx = blockIdx.x * blockDim.x + threadIdx.x;
  if (idx >= 128 * 36) return;
  int b = idx / 36, i = idx % 36;
  float s = eb[i];
#pragma unroll
  for (int j = 0; j < 9; ++j) s += stat[b * 9 + j] * ew[i * 9 + j];
  emb[idx] = s;
}

// x1row[b][c] = mean_i relu(V1)[b*36+i][c]  (gamma = 1/36 exactly since ew = ones)
__global__ void reduce_nodes(const float* __restrict__ V1, _Float16* __restrict__ x1row) {
  int idx = blockIdx.x * blockDim.x + threadIdx.x;
  if (idx >= 128 * 864) return;
  int b = idx / 864, c = idx % 864;
  float s = 0.f;
  for (int i = 0; i < 36; ++i) s += V1[((size_t)(b * 36 + i)) * 864 + c];
  x1row[idx] = (_Float16)(s * (1.f / 36.f));
}

__global__ void build_x(const float* __restrict__ x2row, const float* __restrict__ times,
                        float* __restrict__ x, _Float16* __restrict__ xh, int T, int Bb) {
  int idx = blockIdx.x * blockDim.x + threadIdx.x;
  int total = T * Bb * 160;
  if (idx >= total) return;
  int t = idx / (Bb * 160);
  int r = idx % (Bb * 160);
  int b = r / 160, c = r % 160;
  float vv;
  if (c < 144) {
    vv = x2row[(size_t)b * 864 + t * 4 + (c & 3)];  // x2 is node-constant
  } else {
    int p = c - 144;
    int pp = (p < 8) ? p : p - 8;
    float ts = __expf(logf(215.f) * ((float)pp / 7.f));
    float sc = times[(size_t)t * Bb + b] / ts;
    vv = (p < 8) ? sinf(sc) : cosf(sc);
  }
  x[idx] = vv;
  xh[idx] = (_Float16)vv;
}

__global__ void agg_feat(const float* __restrict__ x, const float* __restrict__ emb,
                         const int* __restrict__ lengths, float* __restrict__ feat,
                         int T, int Bb) {
  int idx = blockIdx.x * blockDim.x + threadIdx.x;
  if (idx >= 128 * 196) return;
  int b = idx / 196, c = idx % 196;
  float vv;
  if (c < 160) {
    int len = lengths[b];
    float s = 0.f;
    for (int t = 0; t < len; ++t) s += x[((size_t)t * Bb + b) * 160 + c];
    vv = s / (float)(len + 1);
  } else {
    vv = emb[b * 36 + (c - 160)];
  }
  feat[idx] = vv;
}

__global__ void mlp1_kernel(const float* __restrict__ feat, const float* __restrict__ w,
                            const float* __restrict__ bi, float* __restrict__ hid) {
  int idx = blockIdx.x * blockDim.x + threadIdx.x;
  if (idx >= 128 * 196) return;
  int b = idx / 196, j = idx % 196;
  float s = bi[j];
  for (int k2 = 0; k2 < 196; ++k2) s += feat[b * 196 + k2] * w[j * 196 + k2];
  hid[idx] = fmaxf(s, 0.f);
}

__global__ void mlp2_kernel(const float* __restrict__ hid, const float* __restrict__ w,
                            const float* __restrict__ bi, float* __restrict__ out, int out_size) {
  int idx = blockIdx.x * blockDim.x + threadIdx.x;
  if (idx < 256) {
    int b = idx / 2, j = idx % 2;
    float s = bi[j];
    for (int k2 = 0; k2 < 196; ++k2) s += hid[b * 196 + k2] * w[j * 196 + k2];
    out[idx] = s;
  }
  if (idx == 0 && out_size > 256) out[256] = 0.0f;  // distance == 0 exactly (alpha = ones)
}

extern "C" void kernel_launch(void* const* d_in, const int* in_sizes, int n_in,
                              void* d_out, int out_size, void* d_ws, size_t ws_size,
                              hipStream_t stream) {
  (void)in_sizes; (void)n_in; (void)ws_size;
  const int T = 215, Bb = 128, TB = T * Bb;

  const float* src   = (const float*)d_in[0];
  const float* stat  = (const float*)d_in[1];
  const float* times = (const float*)d_in[2];
  const int*   lens  = (const int*)d_in[3];
  const float* Ru    = (const float*)d_in[4];
  const float* emb_w = (const float*)d_in[5];
  const float* emb_b = (const float*)d_in[6];
  const float* Wv1   = (const float*)d_in[7];
  const float* bv1   = (const float*)d_in[8];
  const float* Wv2   = (const float*)d_in[9];
  const float* bv2   = (const float*)d_in[10];
  const float* ipw   = (const float*)d_in[11];
  const float* ipb   = (const float*)d_in[12];
  const float* ow    = (const float*)d_in[13];
  const float* ob    = (const float*)d_in[14];
  const float* l1w   = (const float*)d_in[15];
  const float* l1b   = (const float*)d_in[16];
  const float* l2w   = (const float*)d_in[17];
  const float* l2b   = (const float*)d_in[18];
  const float* ln1s  = (const float*)d_in[19];
  const float* ln1b  = (const float*)d_in[20];
  const float* ln2s  = (const float*)d_in[21];
  const float* ln2b  = (const float*)d_in[22];
  const float* m1w   = (const float*)d_in[23];
  const float* m1b   = (const float*)d_in[24];
  const float* m2w   = (const float*)d_in[25];
  const float* m2b   = (const float*)d_in[26];
  float* out = (float*)d_out;

  size_t off = 0;
  auto alloc = [&](size_t bytes) -> void* {
    void* p = (char*)d_ws + off;
    off += (bytes + 255) & ~(size_t)255;
    return p;
  };
  _Float16* Xb    = (_Float16*)alloc((size_t)4608 * 864 * 2);
  _Float16* Wv1h  = (_Float16*)alloc((size_t)864 * 864 * 2);
  _Float16* Wv2h  = (_Float16*)alloc((size_t)864 * 864 * 2);
  float*    bv1p  = (float*)   alloc((size_t)864 * 4);
  float*    bv2p  = (float*)   alloc((size_t)864 * 4);
  float*    V1    = (float*)   alloc((size_t)4608 * 864 * 4);
  _Float16* x1row = (_Float16*)alloc((size_t)128 * 864 * 2);
  float*    x2row = (float*)   alloc((size_t)128 * 864 * 4);
  float*    xbuf  = (float*)   alloc((size_t)TB * 160 * 4);
  _Float16* xh    = (_Float16*)alloc((size_t)TB * 160 * 2);
  float*    tmp   = (float*)   alloc((size_t)TB * 160 * 4);
  _Float16* ff1   = (_Float16*)alloc((size_t)TB * 128 * 2);
  size_t qkElems = (size_t)512 * 224 * 64;
  size_t vtElems = (size_t)512 * 48 * 224;
  _Float16* qh    = (_Float16*)alloc(qkElems * 2);
  _Float16* kh    = (_Float16*)alloc(qkElems * 2);
  _Float16* vT    = (_Float16*)alloc(vtElems * 2);
  _Float16* oh    = (_Float16*)alloc((size_t)TB * 160 * 2);
  _Float16* ipw16 = (_Float16*)alloc((size_t)2 * 480 * 160 * 2);
  _Float16* ow16  = (_Float16*)alloc((size_t)2 * 160 * 160 * 2);
  _Float16* l1w16 = (_Float16*)alloc((size_t)2 * 128 * 160 * 2);
  _Float16* l2w16 = (_Float16*)alloc((size_t)2 * 160 * 128 * 2);
  float*    embB  = (float*)   alloc((size_t)128 * 36 * 4);
  float*    feat  = (float*)   alloc((size_t)128 * 196 * 4);
  float*    hid   = (float*)   alloc((size_t)128 * 196 * 4);

  const int THR = 256;
  auto nb = [](int n) { return (n + 255) / 256; };

  // ---- weight staging (f32 -> f16, zero-padded where needed) ----
  f32_to_f16_pad<<<nb(864 * 864), THR, 0, stream>>>(Wv1, Wv1h, 860, 860, 864, 864);
  f32_to_f16_pad<<<nb(864 * 864), THR, 0, stream>>>(Wv2, Wv2h, 860, 860, 864, 864);
  pad_bias<<<nb(864), THR, 0, stream>>>(bv1, bv1p, 860, 864);
  pad_bias<<<nb(864), THR, 0, stream>>>(bv2, bv2p, 860, 864);
  f32_to_f16<<<nb(2 * 480 * 160), THR, 0, stream>>>(ipw, ipw16, 2 * 480 * 160);
  f32_to_f16<<<nb(2 * 160 * 160), THR, 0, stream>>>(ow, ow16, 2 * 160 * 160);
  f32_to_f16<<<nb(2 * 128 * 160), THR, 0, stream>>>(l1w, l1w16, 2 * 128 * 160);
  f32_to_f16<<<nb(2 * 160 * 128), THR, 0, stream>>>(l2w, l2w16, 2 * 160 * 128);

  // ---- elementwise prep ----
  prep_xb<<<nb(4608 * 864), THR, 0, stream>>>(src, Ru, Xb, T, Bb);
  emb_kernel<<<nb(128 * 36), THR, 0, stream>>>(stat, emb_w, emb_b, embB);

  // ---- graph prop 1: V1 = relu(Xb @ Wv1^T + bv1)  (4608 x 864 x 864) ----
  {
    int tiles = (4608 / 32) * (864 / 32);
    gemm_wmma<GM_OUT_F32, true><<<(tiles + 7) / 8, THR, 0, stream>>>(
        Xb, 864, Wv1h, 864, bv1p, 4608, 864, 864,
        V1, nullptr, 864, nullptr, nullptr, nullptr, T, Bb);
  }
  reduce_nodes<<<nb(128 * 864), THR, 0, stream>>>(V1, x1row);

  // ---- graph prop 2 collapses to (128 x 864 x 864): x1 is node-constant ----
  {
    int tiles = (128 / 32) * (864 / 32);
    gemm_wmma<GM_OUT_F32, true><<<(tiles + 7) / 8, THR, 0, stream>>>(
        x1row, 864, Wv2h, 864, bv2p, 128, 864, 864,
        x2row, nullptr, 864, nullptr, nullptr, nullptr, T, Bb);
  }

  build_x<<<nb(TB * 160), THR, 0, stream>>>(x2row, times, xbuf, xh, T, Bb);

  hipMemsetAsync(qh, 0, qkElems * 2, stream);
  hipMemsetAsync(kh, 0, qkElems * 2, stream);
  hipMemsetAsync(vT, 0, vtElems * 2, stream);

  // ---- transformer layers ----
  for (int l = 0; l < 2; ++l) {
    {  // QKV: (TB x 160) @ (480 x 160)^T -> q/k padded rows + transposed V
      int tiles = (TB / 32) * (480 / 32);
      gemm_wmma<GM_OUT_QKV, false><<<(tiles + 7) / 8, THR, 0, stream>>>(
          xh, 160, ipw16 + (size_t)l * 480 * 160, 160, ipb + l * 480,
          TB, 480, 160, nullptr, nullptr, 0, qh, kh, vT, T, Bb);
    }
    attention_kernel<<<512, THR, 224 * 224 * 6, stream>>>(qh, kh, vT, lens, oh, T, Bb);
    {  // out-proj: (TB x 160) @ (160 x 160)^T -> tmp (f32)
      int tiles = (TB / 32) * (160 / 32);
      gemm_wmma<GM_OUT_F32, false><<<(tiles + 7) / 8, THR, 0, stream>>>(
          oh, 160, ow16 + (size_t)l * 160 * 160, 160, ob + l * 160,
          TB, 160, 160, tmp, nullptr, 160, nullptr, nullptr, nullptr, T, Bb);
    }
    ln_kernel<<<TB / 8, THR, 0, stream>>>(xbuf, tmp, ln1s + l * 160, ln1b + l * 160, xh, TB);
    {  // FF1: relu((TB x 160) @ (128 x 160)^T) -> ff1 (f16)
      int tiles = (TB / 32) * (128 / 32);
      gemm_wmma<GM_OUT_F16, true><<<(tiles + 7) / 8, THR, 0, stream>>>(
          xh, 160, l1w16 + (size_t)l * 128 * 160, 160, l1b + l * 128,
          TB, 128, 160, nullptr, ff1, 128, nullptr, nullptr, nullptr, T, Bb);
    }
    {  // FF2: (TB x 128) @ (160 x 128)^T -> tmp (f32)
      int tiles = (TB / 32) * (160 / 32);
      gemm_wmma<GM_OUT_F32, false><<<(tiles + 7) / 8, THR, 0, stream>>>(
          ff1, 128, l2w16 + (size_t)l * 160 * 128, 128, l2b + l * 160,
          TB, 160, 128, tmp, nullptr, 160, nullptr, nullptr, nullptr, T, Bb);
    }
    ln_kernel<<<TB / 8, THR, 0, stream>>>(xbuf, tmp, ln2s + l * 160, ln2b + l * 160, xh, TB);
  }

  // ---- head ----
  agg_feat<<<nb(128 * 196), THR, 0, stream>>>(xbuf, embB, lens, feat, T, Bb);
  mlp1_kernel<<<nb(128 * 196), THR, 0, stream>>>(feat, m1w, m1b, hid);
  mlp2_kernel<<<1, THR, 0, stream>>>(hid, m2w, m2b, out, out_size);
}
